// ARES_20555713478845
// MI455X (gfx1250) — matrix operand order repaired and checked by reading, not compiled, not run on hardware
//
#include <hip/hip_runtime.h>
#include <stdint.h>

#define EPS 1e-5f

typedef __attribute__((ext_vector_type(16))) __bf16 v16bf;
typedef __attribute__((ext_vector_type(8)))  float  v8f;

__device__ __forceinline__ unsigned short f2bf(float f) {
  union { float f; uint32_t u; } x; x.f = f;
  uint32_t r = x.u + 0x7fffu + ((x.u >> 16) & 1u);
  return (unsigned short)(r >> 16);
}
__device__ __forceinline__ float bf2f(unsigned short h) {
  union { uint32_t u; float f; } x; x.u = ((uint32_t)h) << 16;
  return x.f;
}
__device__ __forceinline__ __bf16 us2bf(unsigned short h) {
  return __builtin_bit_cast(__bf16, h);
}

// load 8 consecutive bf16 (16B, aligned) into fragment elements [e0..e0+7]
__device__ __forceinline__ void load8(__bf16* fe, int e0, const unsigned short* p) {
  const uint4 w = *(const uint4*)p;
  fe[e0+0] = us2bf((unsigned short)(w.x & 0xffffu));
  fe[e0+1] = us2bf((unsigned short)(w.x >> 16));
  fe[e0+2] = us2bf((unsigned short)(w.y & 0xffffu));
  fe[e0+3] = us2bf((unsigned short)(w.y >> 16));
  fe[e0+4] = us2bf((unsigned short)(w.z & 0xffffu));
  fe[e0+5] = us2bf((unsigned short)(w.z >> 16));
  fe[e0+6] = us2bf((unsigned short)(w.w & 0xffffu));
  fe[e0+7] = us2bf((unsigned short)(w.w >> 16));
}
__device__ __forceinline__ void zero8(__bf16* fe, int e0) {
  #pragma unroll
  for (int i = 0; i < 8; ++i) fe[e0+i] = us2bf((unsigned short)0);
}
__device__ __forceinline__ uint4 pack8(const unsigned short* s) {
  uint4 w;
  w.x = (uint32_t)s[0] | ((uint32_t)s[1] << 16);
  w.y = (uint32_t)s[2] | ((uint32_t)s[3] << 16);
  w.z = (uint32_t)s[4] | ((uint32_t)s[5] << 16);
  w.w = (uint32_t)s[6] | ((uint32_t)s[7] << 16);
  return w;
}

#define WMMA_BF16(a, b, c) \
  __builtin_amdgcn_wmma_f32_16x16x32_bf16(false, (a), false, (b), (short)0, (c), false, false)

// Per-lane async global->LDS copy (GLOBAL_LOAD_ASYNC_TO_LDS_B32, tracked by ASYNCcnt).
// ldsoff = workgroup-relative LDS byte offset (low 32 bits of flat shared pointer).
__device__ __forceinline__ void async_g2l_b32(uint32_t ldsoff, const void* gaddr) {
  asm volatile("global_load_async_to_lds_b32 %0, %1, off"
               :: "v"(ldsoff), "v"((uint64_t)(uintptr_t)gaddr) : "memory");
}
__device__ __forceinline__ void wait_asynccnt0() {
  asm volatile("s_wait_asynccnt 0x0" ::: "memory");
}

// ---------------------------------------------------------------------------
// Small prep kernels
// ---------------------------------------------------------------------------
__global__ void f32_to_bf16_kernel(const float* __restrict__ in,
                                   unsigned short* __restrict__ out, size_t n) {
  size_t i = (size_t)blockIdx.x * 256 + threadIdx.x;
  if (i < n) out[i] = f2bf(in[i]);
}

// s = g/sqrt(v+eps); sh = (bconv - m)*s + be (+ extra, used to fold bd into conv2 path)
__global__ void bn_prep_kernel(const float* g, const float* be, const float* m,
                               const float* v, const float* bconv, const float* extra,
                               float* s_out, float* sh_out, int C) {
  int c = blockIdx.x * 256 + threadIdx.x;
  if (c >= C) return;
  float s = g[c] * rsqrtf(v[c] + EPS);
  float sh = (bconv[c] - m[c]) * s + be[c] + (extra ? extra[c] : 0.f);
  s_out[c] = s; sh_out[c] = sh;
}

// Pack conv weights [COUT][CIN][KW] into WMMA A-fragment order, k = t*CIN + ci.
// A 16x32 bf16 order used by this code (B loads use the same k order): lane m=(lane&15); kb = lane<16 ? 0 : 16;
// elements e -> k = kb+e
__global__ void pack_conv_A_kernel(const float* __restrict__ W, unsigned short* __restrict__ out,
                                   int COUT, int CIN, int KW, int nTiles, int nCh) {
  size_t gid = (size_t)blockIdx.x * 256 + threadIdx.x;
  size_t total = (size_t)nTiles * nCh * 512;
  if (gid >= total) return;
  int e    = (int)(gid & 15);
  int lane = (int)((gid >> 4) & 31);
  int ch   = (int)((gid >> 9) % (size_t)nCh);
  int tile = (int)(gid / ((size_t)nCh * 512));
  int co = tile * 16 + (lane & 15);
  int kb = (lane < 16) ? 0 : 16;
  int k = ch * 32 + kb + e;
  float v = 0.f;
  if (co < COUT && k < KW * CIN) {
    int t = k / CIN, ci = k % CIN;
    v = W[((size_t)co * CIN + ci) * KW + t];
  }
  out[gid] = f2bf(v);
}

// Pack linear weights W[N][K] (out = X @ W^T) into WMMA B-fragment order:
// lane n=(lane&15); k = ch*32 + (lane<16 ? 0 : 16) + e  (16 consecutive K/lane)
__global__ void pack_lin_B_kernel(const float* __restrict__ W, unsigned short* __restrict__ out,
                                  int N, int K, int nTiles, int nCh) {
  size_t gid = (size_t)blockIdx.x * 256 + threadIdx.x;
  size_t total = (size_t)nTiles * nCh * 512;
  if (gid >= total) return;
  int e    = (int)(gid & 15);
  int lane = (int)((gid >> 4) & 31);
  int ch   = (int)((gid >> 9) % (size_t)nCh);
  int tile = (int)(gid / ((size_t)nCh * 512));
  int n = tile * 16 + (lane & 15);
  int k = ch * 32 + ((lane < 16) ? 0 : 16) + e;
  float v = (n < N && k < K) ? W[(size_t)n * K + k] : 0.f;
  out[gid] = f2bf(v);
}

// ---------------------------------------------------------------------------
// Fused residual conv block: conv7+BN+ReLU -> conv7+BN + 1x1 res + ReLU -> maxpool(8,4)
// One workgroup = 1 batch elem x 8 pool outputs. All conv1 activations live in LDS.
// ---------------------------------------------------------------------------
template<int CIN, int COUT>
__global__ __launch_bounds__(256)
void conv_block_kernel(const unsigned short* __restrict__ in, int L, int Lp,
                       const unsigned short* __restrict__ w1p,
                       const unsigned short* __restrict__ w2p,
                       const unsigned short* __restrict__ wdp,
                       const float* __restrict__ s1, const float* __restrict__ sh1,
                       const float* __restrict__ s2, const float* __restrict__ sh2,
                       unsigned short* __restrict__ out) {
  __shared__ __align__(16) unsigned short sX [56 * CIN];   // x,  gpos = l0-6 .. l0+49
  __shared__ __align__(16) unsigned short sH1[48 * COUT];  // h1, gpos = l0-3 .. l0+44
  __shared__ __align__(16) unsigned short sH2[48 * COUT];  // h2, gpos = l0   .. l0+47
  const int tid = threadIdx.x, lane = tid & 31, wave = tid >> 5;
  const int b = blockIdx.y;
  const int p0 = blockIdx.x * 8;
  const int l0 = p0 * 4;

  // stage x tile, [pos][ci] layout, zero outside [0,L)
  for (int idx = tid; idx < 56 * CIN; idx += 256) {
    int ci = idx / 56, pos = idx % 56;
    int g = l0 - 6 + pos;
    unsigned short v = 0;
    if (g >= 0 && g < L) v = in[((size_t)b * CIN + ci) * L + g];
    sX[pos * CIN + ci] = v;
  }
  __syncthreads();

  // ---- conv1 + bn1 + relu -> sH1 ----
  {
    const int nCh = (7 * CIN + 31) / 32;
    const int nT = (COUT / 16) * 3;
    for (int f = wave; f < nT; f += 8) {
      int mt = f / 3, nt = f % 3;
      int n = nt * 16 + (lane & 15);        // h1 position index (gpos = l0-3+n)
      v8f acc = {};
      for (int ch = 0; ch < nCh; ++ch) {
        v16bf a = ((const v16bf*)w1p)[(mt * nCh + ch) * 32 + lane];
        v16bf bb; __bf16* fe = (__bf16*)&bb;
        #pragma unroll
        for (int hh = 0; hh < 2; ++hh) {
          int k0 = ch * 32 + ((lane < 16) ? 0 : 16) + hh * 8;
          if (k0 < 7 * CIN) {
            int t = k0 / CIN, ci = k0 % CIN;
            load8(fe, hh * 8, &sX[(n + t) * CIN + ci]);
          } else zero8(fe, hh * 8);
        }
        acc = WMMA_BF16(a, bb, acc);
      }
      const float* ac = (const float*)&acc;
      int g = l0 - 3 + n;
      bool inb = (g >= 0) && (g < L);       // SAME padding: h1 = 0 outside [0,L)
      int co0 = mt * 16 + ((lane < 16) ? 0 : 8);
      unsigned short st[8];
      #pragma unroll
      for (int r = 0; r < 8; ++r) {
        int co = co0 + r;
        float v = inb ? fmaxf(ac[r] * s1[co] + sh1[co], 0.f) : 0.f;
        st[r] = f2bf(v);
      }
      *(uint4*)&sH1[n * COUT + co0] = pack8(st);
    }
  }
  __syncthreads();

  // ---- conv2 + bn2 + 1x1 residual + relu -> sH2 ----
  {
    const int nCh  = (7 * COUT + 31) / 32;
    const int nChR = (CIN + 31) / 32;
    const int nT = (COUT / 16) * 3;
    for (int f = wave; f < nT; f += 8) {
      int mt = f / 3, nt = f % 3;
      int n2 = nt * 16 + (lane & 15);       // output position (gpos = l0+n2); valid n2<40
      v8f acc = {};
      for (int ch = 0; ch < nCh; ++ch) {
        v16bf a = ((const v16bf*)w2p)[(mt * nCh + ch) * 32 + lane];
        v16bf bb; __bf16* fe = (__bf16*)&bb;
        #pragma unroll
        for (int hh = 0; hh < 2; ++hh) {
          int k0 = ch * 32 + ((lane < 16) ? 0 : 16) + hh * 8;
          if (k0 < 7 * COUT) {
            int t = k0 / COUT, co = k0 % COUT;
            int hi = n2 + t; if (hi > 47) hi = 47;   // clamp for pad lanes only
            load8(fe, hh * 8, &sH1[hi * COUT + co]);
          } else zero8(fe, hh * 8);
        }
        acc = WMMA_BF16(a, bb, acc);
      }
      v8f accr = {};
      for (int ch = 0; ch < nChR; ++ch) {
        v16bf a = ((const v16bf*)wdp)[(mt * nChR + ch) * 32 + lane];
        v16bf bb; __bf16* fe = (__bf16*)&bb;
        #pragma unroll
        for (int hh = 0; hh < 2; ++hh) {
          int k0 = ch * 32 + ((lane < 16) ? 0 : 16) + hh * 8;
          if (k0 < CIN) load8(fe, hh * 8, &sX[(n2 + 6) * CIN + k0]);
          else          zero8(fe, hh * 8);
        }
        accr = WMMA_BF16(a, bb, accr);
      }
      const float* ac = (const float*)&acc;
      const float* ar = (const float*)&accr;
      int co0 = mt * 16 + ((lane < 16) ? 0 : 8);
      unsigned short st[8];
      #pragma unroll
      for (int r = 0; r < 8; ++r) {
        int co = co0 + r;
        float v = fmaxf(ac[r] * s2[co] + sh2[co] + ar[r], 0.f);
        st[r] = f2bf(v);
      }
      *(uint4*)&sH2[n2 * COUT + co0] = pack8(st);
    }
  }
  __syncthreads();

  // ---- maxpool k=8 s=4 ----
  for (int idx = tid; idx < 8 * COUT; idx += 256) {
    int pl = idx / COUT, co = idx % COUT;
    int p = p0 + pl;
    if (p >= Lp) continue;
    float m = -3.4e38f;
    #pragma unroll
    for (int i = 0; i < 8; ++i)
      m = fmaxf(m, bf2f(sH2[(4 * pl + i) * COUT + co]));
    out[((size_t)b * COUT + co) * Lp + p] = f2bf(m);
  }
}

// ---------------------------------------------------------------------------
// Generic WMMA GEMM: out[M,N] = act( A[M,K](f32) @ Wpacked^T + bias (+ res) )
// Block tile 64x128 (8 waves x 16x64), LDS-staged bf16 A, K in 32-chunks.
// act: 0=none, 1=exact GELU
// ---------------------------------------------------------------------------
__global__ __launch_bounds__(256)
void gemm_wmma(const float* __restrict__ A, const unsigned short* __restrict__ Bp,
               const float* __restrict__ bias, const float* __restrict__ res,
               float* __restrict__ out, int M, int N, int K, int act) {
  __shared__ __align__(16) unsigned short sA[64 * 32];
  const int tid = threadIdx.x, lane = tid & 31, wave = tid >> 5;
  const int m0 = blockIdx.y * 64;
  const int mf = wave & 3, ng = wave >> 2;
  const int nCh = K >> 5;
  const int tileBase = blockIdx.x * 8 + ng * 4;
  v8f acc[4] = {};
  const v16bf* bp = (const v16bf*)Bp;

  const int fr = tid >> 2;        // 0..63
  const int fc = (tid & 3) * 8;   // 0,8,16,24
  for (int ch = 0; ch < nCh; ++ch) {
    {
      const float* src = A + (size_t)(m0 + fr) * K + ch * 32 + fc;
      if (ch + 1 < nCh) __builtin_prefetch(src + 32, 0, 1);  // global_prefetch_b8
      unsigned short tmp[8];
      #pragma unroll
      for (int i = 0; i < 8; ++i) tmp[i] = f2bf(src[i]);
      *(uint4*)&sA[fr * 32 + fc] = pack8(tmp);
    }
    __syncthreads();
    v16bf a; __bf16* ae = (__bf16*)&a;
    int mr = mf * 16 + (lane & 15);
    int kb = (lane < 16) ? 0 : 16;
    load8(ae, 0, &sA[mr * 32 + kb]);
    load8(ae, 8, &sA[mr * 32 + kb + 8]);
    #pragma unroll
    for (int j = 0; j < 4; ++j) {
      v16bf bb = bp[((size_t)(tileBase + j) * nCh + ch) * 32 + lane];
      acc[j] = WMMA_BF16(a, bb, acc[j]);
    }
    __syncthreads();
  }
  int mr0 = m0 + mf * 16 + ((lane < 16) ? 0 : 8);
  #pragma unroll
  for (int j = 0; j < 4; ++j) {
    int n = (tileBase + j) * 16 + (lane & 15);
    if (n >= N) continue;
    float bv = bias ? bias[n] : 0.f;
    const float* ac = (const float*)&acc[j];
    #pragma unroll
    for (int r = 0; r < 8; ++r) {
      int m = mr0 + r;
      float v = ac[r] + bv;
      if (res) v += res[(size_t)m * N + n];
      if (act == 1) v = 0.5f * v * (1.f + erff(v * 0.70710678118654752f));
      out[(size_t)m * N + n] = v;
    }
  }
}

// ---------------------------------------------------------------------------
// LayerNorm over C=256; one wave per row
// ---------------------------------------------------------------------------
__global__ __launch_bounds__(256)
void layernorm_kernel(const float* __restrict__ X, const float* __restrict__ g,
                      const float* __restrict__ bta, float* __restrict__ Y, int M) {
  const int lane = threadIdx.x & 31, wave = threadIdx.x >> 5;
  const int row = blockIdx.x * 8 + wave;
  if (row >= M) return;
  const float* x = X + (size_t)row * 256;
  float s = 0.f, s2 = 0.f;
  #pragma unroll
  for (int i = 0; i < 8; ++i) { float v = x[lane + i * 32]; s += v; s2 += v * v; }
  #pragma unroll
  for (int off = 16; off; off >>= 1) {
    s  += __shfl_xor(s,  off, 32);
    s2 += __shfl_xor(s2, off, 32);
  }
  float mu = s * (1.f / 256.f);
  float var = s2 * (1.f / 256.f) - mu * mu;
  float inv = rsqrtf(var + EPS);
  float* y = Y + (size_t)row * 256;
  #pragma unroll
  for (int i = 0; i < 8; ++i) {
    int c = lane + i * 32;
    y[c] = (x[c] - mu) * inv * g[c] + bta[c];
  }
}

// ---------------------------------------------------------------------------
// Top-20 attention, N=29, D=32, one wave32 per (batch, head).
// Q/K/V staged global->LDS with GLOBAL_LOAD_ASYNC_TO_LDS_B32 (ASYNCcnt path).
// ---------------------------------------------------------------------------
__global__ __launch_bounds__(32)
void topm_attn_kernel(const float* __restrict__ QKV, float* __restrict__ OUT) {
  const int lane = threadIdx.x;
  const int b = blockIdx.x >> 3;
  const int h = blockIdx.x & 7;
  __shared__ __align__(16) float sQ[29 * 32];
  __shared__ __align__(16) float sK[29 * 32];
  __shared__ __align__(16) float sV[29 * 32];
  __shared__ float sP[32];
  const size_t base = (size_t)b * 29 * 768 + (size_t)h * 32;

  // async per-lane copies: one f32 per lane per row per matrix
  const uint32_t oQ = (uint32_t)(uintptr_t)(void*)sQ;
  const uint32_t oK = (uint32_t)(uintptr_t)(void*)sK;
  const uint32_t oV = (uint32_t)(uintptr_t)(void*)sV;
  const float* gq = QKV + base;
  for (int t = 0; t < 29; ++t) {
    uint32_t d = (uint32_t)(t * 32 + lane) * 4u;
    const float* pq = gq + (size_t)t * 768 + lane;
    async_g2l_b32(oQ + d, pq);
    async_g2l_b32(oK + d, pq + 256);
    async_g2l_b32(oV + d, pq + 512);
  }
  wait_asynccnt0();
  __syncthreads();

  const float scale = 0.17677669529663687f;  // 32^-0.5
  for (int t = 0; t < 29; ++t) {
    float s = -3.4e38f;
    bool active = lane < 29;
    if (active) {
      float d = 0.f;
      #pragma unroll
      for (int k = 0; k < 32; ++k) d += sQ[t * 32 + k] * sK[lane * 32 + k];
      s = d * scale;
    }
    // drop the 9 smallest of 29 -> exact top-20 set
    for (int it = 0; it < 9; ++it) {
      float v = active ? s : 3.4e38f;
      float mn = v;
      #pragma unroll
      for (int off = 16; off; off >>= 1) mn = fminf(mn, __shfl_xor(mn, off, 32));
      uint32_t msk = __builtin_amdgcn_ballot_w32(active && (s == mn));
      int leader = (int)__builtin_ctz(msk);
      if (lane == leader) active = false;
    }
    float v = active ? s : -3.4e38f;
    float mx = v;
    #pragma unroll
    for (int off = 16; off; off >>= 1) mx = fmaxf(mx, __shfl_xor(mx, off, 32));
    float e = active ? __expf(s - mx) : 0.f;
    float sum = e;
    #pragma unroll
    for (int off = 16; off; off >>= 1) sum += __shfl_xor(sum, off, 32);
    sP[lane] = e / sum;
    __syncthreads();
    float o = 0.f;
    for (int j = 0; j < 29; ++j) o += sP[j] * sV[j * 32 + lane];
    OUT[((size_t)b * 29 + t) * 256 + (size_t)h * 32 + lane] = o;
    __syncthreads();
  }
}

// [B,256,29] bf16 -> [B,29,256] f32 + pos_embed
__global__ void transpose_pos_kernel(const unsigned short* __restrict__ act,
                                     const float* __restrict__ pos, float* __restrict__ X) {
  size_t gid = (size_t)blockIdx.x * 256 + threadIdx.x;
  if (gid >= (size_t)256 * 29 * 256) return;
  int c = (int)(gid & 255);
  int t = (int)((gid >> 8) % 29);
  int b = (int)(gid / (29 * 256));
  X[gid] = bf2f(act[((size_t)b * 256 + c) * 29 + t]) + pos[t * 256 + c];
}

__global__ void meanpool_kernel(const float* __restrict__ X, float* __restrict__ P) {
  int gid = blockIdx.x * 256 + threadIdx.x;
  if (gid >= 256 * 256) return;
  int c = gid & 255, b = gid >> 8;
  float s = 0.f;
  for (int t = 0; t < 29; ++t) s += X[((size_t)b * 29 + t) * 256 + c];
  P[gid] = s * (1.f / 29.f);
}

// ---------------------------------------------------------------------------
// Orchestration
// ---------------------------------------------------------------------------
extern "C" void kernel_launch(void* const* d_in, const int* in_sizes, int n_in,
                              void* d_out, int out_size, void* d_ws, size_t ws_size,
                              hipStream_t stream) {
  (void)in_sizes; (void)n_in; (void)out_size; (void)ws_size;
  const int B = 256;
  const float* x_in   = (const float*)d_in[0];
  const float* pos    = (const float*)d_in[57];
  const float* head_w = (const float*)d_in[106];
  const float* head_b = (const float*)d_in[107];

  char* wsb = (char*)d_ws;
  size_t cur = 0;
  auto alloc = [&](size_t bytes) -> void* {
    void* p = wsb + cur;
    cur += (bytes + 255) & ~(size_t)255;
    return p;
  };

  const int cinA[4]  = {8, 32, 64, 128};
  const int coutA[4] = {32, 64, 128, 256};
  const int LA[4]    = {8000, 1999, 498, 123};
  const int LpA[4]   = {1999, 498, 123, 29};

  unsigned short* act0 = (unsigned short*)alloc((size_t)B * 8 * 8000 * 2);
  unsigned short* actP[4];
  for (int i = 0; i < 4; ++i)
    actP[i] = (unsigned short*)alloc((size_t)B * coutA[i] * LpA[i] * 2);

  unsigned short *w1p[4], *w2p[4], *wdp[4];
  float *s1v[4], *sh1v[4], *s2v[4], *sh2v[4];
  int nCh1A[4], nCh2A[4], nChRA[4];
  for (int i = 0; i < 4; ++i) {
    int nT = coutA[i] / 16;
    nCh1A[i] = (7 * cinA[i]  + 31) / 32;
    nCh2A[i] = (7 * coutA[i] + 31) / 32;
    nChRA[i] = (cinA[i] + 31) / 32;
    w1p[i] = (unsigned short*)alloc((size_t)nT * nCh1A[i] * 512 * 2);
    w2p[i] = (unsigned short*)alloc((size_t)nT * nCh2A[i] * 512 * 2);
    wdp[i] = (unsigned short*)alloc((size_t)nT * nChRA[i] * 512 * 2);
    s1v[i]  = (float*)alloc((size_t)coutA[i] * 4);
    sh1v[i] = (float*)alloc((size_t)coutA[i] * 4);
    s2v[i]  = (float*)alloc((size_t)coutA[i] * 4);
    sh2v[i] = (float*)alloc((size_t)coutA[i] * 4);
  }

  const int MT = 256 * 29;  // 7424 tokens
  float* X   = (float*)alloc((size_t)MT * 256 * 4);
  float* X2  = (float*)alloc((size_t)MT * 256 * 4);
  float* Y   = (float*)alloc((size_t)MT * 256 * 4);
  float* QKV = (float*)alloc((size_t)MT * 768 * 4);
  float* ATT = (float*)alloc((size_t)MT * 256 * 4);
  float* H1  = (float*)alloc((size_t)MT * 1024 * 4);
  float* P   = (float*)alloc((size_t)256 * 256 * 4);

  const int tQKV = ((768 + 127) / 128) * 8;
  const int tPROJ = ((256 + 127) / 128) * 8;
  const int tFC1 = ((1024 + 127) / 128) * 8;
  const int tFC2 = ((256 + 127) / 128) * 8;
  const int tHEAD = ((100 + 127) / 128) * 8;
  unsigned short *qkvp[4], *projp[4], *fc1p[4], *fc2p[4];
  for (int l = 0; l < 4; ++l) {
    qkvp[l]  = (unsigned short*)alloc((size_t)tQKV * 8 * 512 * 2);
    projp[l] = (unsigned short*)alloc((size_t)tPROJ * 8 * 512 * 2);
    fc1p[l]  = (unsigned short*)alloc((size_t)tFC1 * 8 * 512 * 2);
    fc2p[l]  = (unsigned short*)alloc((size_t)tFC2 * 32 * 512 * 2);
  }
  unsigned short* headp = (unsigned short*)alloc((size_t)tHEAD * 8 * 512 * 2);

  // 1) input f32 -> bf16
  {
    size_t n = (size_t)B * 8 * 8000;
    f32_to_bf16_kernel<<<(unsigned)((n + 255) / 256), 256, 0, stream>>>(x_in, act0, n);
  }

  // 2) conv weight packs + BN folds
  for (int i = 0; i < 4; ++i) {
    int base = 1 + i * 14;
    const float* w1 = (const float*)d_in[base + 0];
    const float* b1 = (const float*)d_in[base + 1];
    const float* g1 = (const float*)d_in[base + 2];
    const float* be1 = (const float*)d_in[base + 3];
    const float* m1 = (const float*)d_in[base + 4];
    const float* v1 = (const float*)d_in[base + 5];
    const float* w2 = (const float*)d_in[base + 6];
    const float* b2 = (const float*)d_in[base + 7];
    const float* g2 = (const float*)d_in[base + 8];
    const float* be2 = (const float*)d_in[base + 9];
    const float* m2 = (const float*)d_in[base + 10];
    const float* v2 = (const float*)d_in[base + 11];
    const float* wd = (const float*)d_in[base + 12];
    const float* bd = (const float*)d_in[base + 13];
    int nT = coutA[i] / 16;
    size_t n1 = (size_t)nT * nCh1A[i] * 512;
    size_t n2 = (size_t)nT * nCh2A[i] * 512;
    size_t nr = (size_t)nT * nChRA[i] * 512;
    pack_conv_A_kernel<<<(unsigned)((n1 + 255) / 256), 256, 0, stream>>>(
        w1, w1p[i], coutA[i], cinA[i], 7, nT, nCh1A[i]);
    pack_conv_A_kernel<<<(unsigned)((n2 + 255) / 256), 256, 0, stream>>>(
        w2, w2p[i], coutA[i], coutA[i], 7, nT, nCh2A[i]);
    pack_conv_A_kernel<<<(unsigned)((nr + 255) / 256), 256, 0, stream>>>(
        wd, wdp[i], coutA[i], cinA[i], 1, nT, nChRA[i]);
    bn_prep_kernel<<<1, 256, 0, stream>>>(g1, be1, m1, v1, b1, nullptr, s1v[i], sh1v[i], coutA[i]);
    bn_prep_kernel<<<1, 256, 0, stream>>>(g2, be2, m2, v2, b2, bd, s2v[i], sh2v[i], coutA[i]);
  }

  // 3) fused conv blocks
  conv_block_kernel<8, 32><<<dim3((LpA[0] + 7) / 8, B), 256, 0, stream>>>(
      act0, LA[0], LpA[0], w1p[0], w2p[0], wdp[0], s1v[0], sh1v[0], s2v[0], sh2v[0], actP[0]);
  conv_block_kernel<32, 64><<<dim3((LpA[1] + 7) / 8, B), 256, 0, stream>>>(
      actP[0], LA[1], LpA[1], w1p[1], w2p[1], wdp[1], s1v[1], sh1v[1], s2v[1], sh2v[1], actP[1]);
  conv_block_kernel<64, 128><<<dim3((LpA[2] + 7) / 8, B), 256, 0, stream>>>(
      actP[1], LA[2], LpA[2], w1p[2], w2p[2], wdp[2], s1v[2], sh1v[2], s2v[2], sh2v[2], actP[2]);
  conv_block_kernel<128, 256><<<dim3((LpA[3] + 7) / 8, B), 256, 0, stream>>>(
      actP[2], LA[3], LpA[3], w1p[3], w2p[3], wdp[3], s1v[3], sh1v[3], s2v[3], sh2v[3], actP[3]);

  // 4) transpose + pos embed
  {
    size_t n = (size_t)B * 29 * 256;
    transpose_pos_kernel<<<(unsigned)((n + 255) / 256), 256, 0, stream>>>(actP[3], pos, X);
  }

  // 5) pack linear weights
  for (int l = 0; l < 4; ++l) {
    int base = 58 + l * 12;
    size_t n;
    n = (size_t)tQKV * 8 * 512;
    pack_lin_B_kernel<<<(unsigned)((n + 255) / 256), 256, 0, stream>>>(
        (const float*)d_in[base + 0], qkvp[l], 768, 256, tQKV, 8);
    n = (size_t)tPROJ * 8 * 512;
    pack_lin_B_kernel<<<(unsigned)((n + 255) / 256), 256, 0, stream>>>(
        (const float*)d_in[base + 2], projp[l], 256, 256, tPROJ, 8);
    n = (size_t)tFC1 * 8 * 512;
    pack_lin_B_kernel<<<(unsigned)((n + 255) / 256), 256, 0, stream>>>(
        (const float*)d_in[base + 8], fc1p[l], 1024, 256, tFC1, 8);
    n = (size_t)tFC2 * 32 * 512;
    pack_lin_B_kernel<<<(unsigned)((n + 255) / 256), 256, 0, stream>>>(
        (const float*)d_in[base + 10], fc2p[l], 256, 1024, tFC2, 32);
  }
  {
    size_t n = (size_t)tHEAD * 8 * 512;
    pack_lin_B_kernel<<<(unsigned)((n + 255) / 256), 256, 0, stream>>>(
        head_w, headp, 100, 256, tHEAD, 8);
  }

  // 6) transformer layers
  for (int l = 0; l < 4; ++l) {
    int base = 58 + l * 12;
    const float* qkv_b = (const float*)d_in[base + 1];
    const float* proj_b = (const float*)d_in[base + 3];
    const float* ln1_g = (const float*)d_in[base + 4];
    const float* ln1_b = (const float*)d_in[base + 5];
    const float* ln2_g = (const float*)d_in[base + 6];
    const float* ln2_b = (const float*)d_in[base + 7];
    const float* fc1_b = (const float*)d_in[base + 9];
    const float* fc2_b = (const float*)d_in[base + 11];

    layernorm_kernel<<<MT / 8, 256, 0, stream>>>(X, ln1_g, ln1_b, Y, MT);
    gemm_wmma<<<dim3(6, MT / 64), 256, 0, stream>>>(Y, qkvp[l], qkv_b, nullptr, QKV,
                                                    MT, 768, 256, 0);
    topm_attn_kernel<<<B * 8, 32, 0, stream>>>(QKV, ATT);
    gemm_wmma<<<dim3(2, MT / 64), 256, 0, stream>>>(ATT, projp[l], proj_b, X, X2,
                                                    MT, 256, 256, 0);
    layernorm_kernel<<<MT / 8, 256, 0, stream>>>(X2, ln2_g, ln2_b, Y, MT);
    gemm_wmma<<<dim3(8, MT / 64), 256, 0, stream>>>(Y, fc1p[l], fc1_b, nullptr, H1,
                                                    MT, 1024, 256, 1);
    gemm_wmma<<<dim3(2, MT / 64), 256, 0, stream>>>(H1, fc2p[l], fc2_b, X2, X,
                                                    MT, 256, 1024, 0);
  }

  // 7) mean pool + head
  meanpool_kernel<<<256, 256, 0, stream>>>(X, P);
  gemm_wmma<<<dim3(1, 256 / 64), 256, 0, stream>>>(P, headp, head_b, nullptr,
                                                   (float*)d_out, 256, 100, 256, 0);
}
